// SequenceLoss_35716948033830
// MI455X (gfx1250) — matrix operand order repaired and checked by reading, not compiled, not run on hardware
//
#include <hip/hip_runtime.h>

typedef __attribute__((ext_vector_type(16))) _Float16 v16h;
typedef __attribute__((ext_vector_type(8)))  _Float16 v8h;
typedef __attribute__((ext_vector_type(8)))  float    v8f;
typedef __attribute__((ext_vector_type(4)))  float    v4f;

#define BB 16
#define NN 4096
#define MM 2048
#define DD 256
#define NSPLIT 2            // independent N-range partial-max planes
#define NTILES (NN / NSPLIT / 16)

// ---------------------------------------------------------------------------
// Pass 1: L2-normalize rows of x1 and x2 (fp32 math), emit f16 rows into ws.
// One wave (32 lanes) per row of 256 floats -> 8 floats/lane.
// ---------------------------------------------------------------------------
__global__ __launch_bounds__(256) void norm_kernel(
    const float* __restrict__ x1, const float* __restrict__ x2,
    _Float16* __restrict__ aH, _Float16* __restrict__ bH) {
  const int wave = threadIdx.x >> 5;
  const int lane = threadIdx.x & 31;
  const long row = (long)blockIdx.x * 8 + wave;

  const float* src;
  _Float16* dst;
  if (row < (long)BB * NN) {
    src = x1 + row * DD;
    dst = aH + row * DD;
  } else {
    const long r2 = row - (long)BB * NN;
    src = x2 + r2 * DD;
    dst = bH + r2 * DD;
  }

  v4f v0 = *(const v4f*)(src + lane * 8);
  v4f v1 = *(const v4f*)(src + lane * 8 + 4);
  float ss = 0.f;
#pragma unroll
  for (int i = 0; i < 4; ++i) ss += v0[i] * v0[i] + v1[i] * v1[i];
#pragma unroll
  for (int off = 16; off > 0; off >>= 1) ss += __shfl_xor(ss, off, 32);

  const float inv = 1.0f / fmaxf(sqrtf(ss), 1e-12f);

  v8h h;
#pragma unroll
  for (int i = 0; i < 4; ++i) {
    h[i]     = (_Float16)(v0[i] * inv);
    h[4 + i] = (_Float16)(v1[i] * inv);
  }
  *(v8h*)(dst + lane * 8) = h;
}

// ---------------------------------------------------------------------------
// Pass 2: cos = aH @ bH^T with running max over an N sub-range.
// Wave tile: 16 rows x 32 cols (two resident B fragment sets -> each A
// fragment feeds two WMMAs). 8 waves/WG own 256 columns; grid covers
// NSPLIT x BB x (MM/256). Partial column maxes land in maxs[ns][b][m].
// ---------------------------------------------------------------------------
__global__ __launch_bounds__(256) void simmax_kernel(
    const _Float16* __restrict__ aH, const _Float16* __restrict__ bH,
    float* __restrict__ maxs) {
  const int wave = threadIdx.x >> 5;
  const int lane = threadIdx.x & 31;
  const int idx  = blockIdx.x;
  const int mblk = idx & 7;          // MM/256 == 8 column blocks per batch
  const int b    = (idx >> 3) & 15;  // batch
  const int ns   = idx >> 7;         // N split id
  const int m0   = mblk * 256 + wave * 32;   // first of this wave's 32 cols
  const int col  = lane & 15;
  const int hi   = lane >> 4;
  const int n0   = ns * (NN / NSPLIT);

  // B fragment, contiguous k order used here (same as A): lanes 0-15 hold K=0..15 of column
  // `lane`, lanes 16-31 hold K=16..31 of column `lane-16` -> one contiguous
  // 32-byte run per lane per K-chunk. Two column-tiles resident per wave.
  v16h bfragA[8], bfragB[8];
#pragma unroll
  for (int c = 0; c < 8; ++c) {
    const _Float16* p0 =
        bH + ((long)b * MM + m0 + col) * DD + hi * 16 + c * 32;
    const _Float16* p1 = p0 + 16 * DD;  // cols m0+16 .. m0+31
    v8h l0 = *(const v8h*)(p0);
    v8h h0 = *(const v8h*)(p0 + 8);
    v8h l1 = *(const v8h*)(p1);
    v8h h1 = *(const v8h*)(p1 + 8);
    v16h fA, fB;
#pragma unroll
    for (int i = 0; i < 8; ++i) {
      fA[i] = l0[i]; fA[8 + i] = h0[i];
      fB[i] = l1[i]; fB[8 + i] = h1[i];
    }
    bfragA[c] = fA;
    bfragB[c] = fB;
  }

  float runmaxA = -2.0f, runmaxB = -2.0f;  // cosines live in [-1, 1]

  for (int nt = 0; nt < NTILES; ++nt) {
    // A fragment, contiguous k order used here (same as B): lane holds row lane%16 with one
    // contiguous 16-half run: K = d0+hi*16..+15. Load the whole
    // 16x256 tile (8 fragments) up front so the scheduler can clause all
    // 16 b128 loads and drain them across the 16-WMMA chain below.
    const _Float16* arow =
        aH + ((long)b * NN + n0 + nt * 16 + col) * DD + hi * 16;
    v16h afrag[8];
#pragma unroll
    for (int c = 0; c < 8; ++c) {
      v8h r0 = *(const v8h*)(arow + c * 32);
      v8h r1 = *(const v8h*)(arow + c * 32 + 8);
      v16h a;
#pragma unroll
      for (int i = 0; i < 8; ++i) { a[i] = r0[i]; a[8 + i] = r1[i]; }
      afrag[c] = a;
    }

    v8f accA = {}, accB = {};
#pragma unroll
    for (int c = 0; c < 8; ++c) {
      accA = __builtin_amdgcn_wmma_f32_16x16x32_f16(
          false, afrag[c], false, bfragA[c], (short)0, accA, false, false);
      accB = __builtin_amdgcn_wmma_f32_16x16x32_f16(
          false, afrag[c], false, bfragB[c], (short)0, accB, false, false);
    }

    // C/D layout: lane holds column lane%16, rows M = hi*8 + (0..7) across
    // the 8 accumulator VGPRs -> per-lane max folds 8 rows of this column.
    float tA = accA[0], tB = accB[0];
#pragma unroll
    for (int i = 1; i < 8; ++i) {
      tA = fmaxf(tA, accA[i]);
      tB = fmaxf(tB, accB[i]);
    }
    runmaxA = fmaxf(runmaxA, tA);
    runmaxB = fmaxf(runmaxB, tB);
  }

  // Combine the two row-halves (lanes l and l+16 share the same column).
  runmaxA = fmaxf(runmaxA, __shfl_xor(runmaxA, 16, 32));
  runmaxB = fmaxf(runmaxB, __shfl_xor(runmaxB, 16, 32));
  if (lane < 16) {
    float* base = maxs + ((long)ns * BB + b) * MM + m0;
    base[lane]      = runmaxA;
    base[16 + lane] = runmaxB;
  }
}

// ---------------------------------------------------------------------------
// Pass 3: fold NSPLIT partial-max planes, per-batch mean over M, squared
// error vs y.
// ---------------------------------------------------------------------------
__global__ __launch_bounds__(256) void loss_kernel(
    const float* __restrict__ maxs, const float* __restrict__ y,
    float* __restrict__ batch_loss) {
  const int b = blockIdx.x;
  float s = 0.f;
  for (int i = threadIdx.x; i < MM; i += 256) {
    float m = maxs[(long)b * MM + i];
#pragma unroll
    for (int ns = 1; ns < NSPLIT; ++ns)
      m = fmaxf(m, maxs[((long)ns * BB + b) * MM + i]);
    s += m;
  }
#pragma unroll
  for (int off = 16; off > 0; off >>= 1) s += __shfl_xor(s, off, 32);

  __shared__ float wsum[8];
  if ((threadIdx.x & 31) == 0) wsum[threadIdx.x >> 5] = s;
  __syncthreads();
  if (threadIdx.x == 0) {
    float t = 0.f;
#pragma unroll
    for (int i = 0; i < 8; ++i) t += wsum[i];
    const float mean  = t / (float)MM;
    const float yb    = y[b];
    const float d     = mean - yb;
    const float scale = (yb != 0.0f) ? 1.0f /* Y_SCALE */ : 1.0f;
    batch_loss[b] = d * d * scale;
  }
}

__global__ void final_kernel(const float* __restrict__ batch_loss,
                             float* __restrict__ out) {
  if (threadIdx.x == 0) {
    float s = 0.f;
#pragma unroll
    for (int i = 0; i < BB; ++i) s += batch_loss[i];
    out[0] = s;
  }
}

// ---------------------------------------------------------------------------
extern "C" void kernel_launch(void* const* d_in, const int* in_sizes, int n_in,
                              void* d_out, int out_size, void* d_ws, size_t ws_size,
                              hipStream_t stream) {
  (void)in_sizes; (void)n_in; (void)out_size; (void)ws_size;

  const float* x1 = (const float*)d_in[0];
  const float* x2 = (const float*)d_in[1];
  const float* y  = (const float*)d_in[2];
  float* out = (float*)d_out;

  char* ws = (char*)d_ws;
  _Float16* aH   = (_Float16*)(ws);                               // 32 MiB
  _Float16* bH   = (_Float16*)(ws + (size_t)BB * NN * DD * 2);    // 16 MiB
  float*    maxs = (float*)(ws + (size_t)BB * (NN + MM) * DD * 2);// 256 KiB
  float*    bl   = (float*)((char*)maxs + (size_t)NSPLIT * BB * MM * 4);

  const int rows = BB * NN + BB * MM;                 // 98304 rows
  norm_kernel<<<rows / 8, 256, 0, stream>>>(x1, x2, aH, bH);
  simmax_kernel<<<NSPLIT * BB * (MM / 256), 256, 0, stream>>>(aH, bH, maxs);
  loss_kernel<<<BB, 256, 0, stream>>>(maxs, y, bl);
  final_kernel<<<1, 64, 0, stream>>>(bl, out);
}
